// Qwen3MoEBlock_44418551775993
// MI455X (gfx1250) — matrix operand order repaired and checked by reading, not compiled, not run on hardware
//
#include <hip/hip_runtime.h>
#include <hip/hip_bf16.h>
#include <stdint.h>

#define T_TOKENS 8192
#define HIDDEN   1024
#define INTER    512
#define NEXP     8

typedef __attribute__((ext_vector_type(16))) __bf16       v16bf;
typedef __attribute__((ext_vector_type(8)))  float        v8f;
typedef __attribute__((ext_vector_type(4)))  unsigned int u32x4;
typedef __attribute__((ext_vector_type(4)))  float        f32x4;
typedef int v4i __attribute__((__vector_size__(16)));

// ---- feature probes ------------------------------------------------------
#if defined(__has_builtin)
#if __has_builtin(__builtin_amdgcn_global_load_async_to_lds_b128)
#define HAVE_ASYNC_LDS 1
#endif
#endif
#ifndef HAVE_ASYNC_LDS
#define HAVE_ASYNC_LDS 0
#endif

#if defined(__has_builtin) && __has_builtin(__builtin_amdgcn_rcpf)
#define FAST_RCP(x) __builtin_amdgcn_rcpf(x)
#else
#define FAST_RCP(x) (1.0f / (x))
#endif

#if HAVE_ASYNC_LDS
typedef __attribute__((address_space(1))) v4i gv4i;   // global (AS1) 16B vector
typedef __attribute__((address_space(3))) v4i lv4i;   // LDS (AS3) 16B vector
#endif

// Copy 16 bytes global -> LDS. Async path via gfx1250 builtin when available;
// plain LDS store fallback otherwise.
__device__ __forceinline__ void copy16_g2l(const unsigned short* g, unsigned short* l) {
#if HAVE_ASYNC_LDS
  // generic LDS pointer low 32 bits == LDS byte address (aperture layout);
  // generic global pointer value == global address.
  __builtin_amdgcn_global_load_async_to_lds_b128(
      (gv4i*)(unsigned long long)g,
      (lv4i*)(unsigned int)(unsigned long long)l,
      0, 0);
#else
  *reinterpret_cast<u32x4*>(l) = *reinterpret_cast<const u32x4*>(g);
#endif
}

__device__ __forceinline__ void wait_async_all() {
#if HAVE_ASYNC_LDS
#if defined(__has_builtin) && __has_builtin(__builtin_amdgcn_s_wait_asynccnt)
  __builtin_amdgcn_s_wait_asynccnt(0);
#else
  asm volatile("s_wait_asynccnt 0x0" ::: "memory");
#endif
#endif
}

__device__ __forceinline__ unsigned short f32_to_bf16_rne(float f) {
  union { float f; uint32_t u; } v; v.f = f;
  uint32_t u = v.u;
  u += 0x7FFFu + ((u >> 16) & 1u);   // round-to-nearest-even
  return (unsigned short)(u >> 16);
}

__device__ __forceinline__ v16bf load_frag_lds(const unsigned short* p0,
                                               const unsigned short* p1) {
  v16bf r;
  u32x4* d = reinterpret_cast<u32x4*>(&r);
  d[0] = *reinterpret_cast<const u32x4*>(p0);
  d[1] = *reinterpret_cast<const u32x4*>(p1);
  return r;
}

#define WMMA_BF16(a, b, c) \
  __builtin_amdgcn_wmma_f32_16x16x32_bf16(false, (a), false, (b), (short)0, (c), false, false)

// ---------------------------------------------------------------------------
__global__ void zero_cnt_kernel(int* cnt) {
  if (threadIdx.x < NEXP * 2) cnt[threadIdx.x] = 0;
}

// ---------------------------------------------------------------------------
// Router: one wave per token. logits -> softmax -> top-2 -> renorm.
// ---------------------------------------------------------------------------
__global__ __launch_bounds__(256) void router_kernel(
    const float* __restrict__ x, const float* __restrict__ Wgate,
    float* __restrict__ logits_out, int* __restrict__ cnt,
    int* __restrict__ toklist, float* __restrict__ wtlist) {
  const int wave = threadIdx.x >> 5;
  const int lane = threadIdx.x & 31;
  const int t = blockIdx.x * 8 + wave;
  const float* xr = x + (size_t)t * HIDDEN;

  float acc[NEXP];
#pragma unroll
  for (int e = 0; e < NEXP; ++e) acc[e] = 0.0f;

  for (int h = lane; h < HIDDEN; h += 32) {
    float xv = xr[h];
    const f32x4* wrow = reinterpret_cast<const f32x4*>(Wgate + (size_t)h * NEXP);
    f32x4 w0 = wrow[0], w1 = wrow[1];
    acc[0] += xv * w0.x; acc[1] += xv * w0.y; acc[2] += xv * w0.z; acc[3] += xv * w0.w;
    acc[4] += xv * w1.x; acc[5] += xv * w1.y; acc[6] += xv * w1.z; acc[7] += xv * w1.w;
  }
#pragma unroll
  for (int off = 16; off >= 1; off >>= 1)
#pragma unroll
    for (int e = 0; e < NEXP; ++e) acc[e] += __shfl_down(acc[e], off, 32);

  if (lane == 0) {
    float mx = acc[0];
#pragma unroll
    for (int e = 1; e < NEXP; ++e) mx = fmaxf(mx, acc[e]);
    float p[NEXP], s = 0.0f;
#pragma unroll
    for (int e = 0; e < NEXP; ++e) { p[e] = __expf(acc[e] - mx); s += p[e]; }
    float inv = 1.0f / s;
#pragma unroll
    for (int e = 0; e < NEXP; ++e) p[e] *= inv;

    int e0 = 0;
#pragma unroll
    for (int e = 1; e < NEXP; ++e) if (p[e] > p[e0]) e0 = e;   // first index wins ties
    int e1 = -1;
#pragma unroll
    for (int e = 0; e < NEXP; ++e) {
      if (e == e0) continue;
      if (e1 < 0 || p[e] > p[e1]) e1 = e;
    }

    float w0 = p[e0], w1 = p[e1];
    float swi = 1.0f / (w0 + w1);
    w0 *= swi; w1 *= swi;

#pragma unroll
    for (int e = 0; e < NEXP; ++e) logits_out[(size_t)t * NEXP + e] = acc[e];

    int pos0 = atomicAdd(&cnt[e0 * 2 + 0], 1);
    toklist[(size_t)(e0 * 2 + 0) * T_TOKENS + pos0] = t;
    wtlist [(size_t)(e0 * 2 + 0) * T_TOKENS + pos0] = w0;
    int pos1 = atomicAdd(&cnt[e1 * 2 + 1], 1);
    toklist[(size_t)(e1 * 2 + 1) * T_TOKENS + pos1] = t;
    wtlist [(size_t)(e1 * 2 + 1) * T_TOKENS + pos1] = w1;
  }
}

// ---------------------------------------------------------------------------
// x (f32) -> xb (bf16), 4 per thread
// ---------------------------------------------------------------------------
__global__ __launch_bounds__(256) void convert_x_kernel(const float* __restrict__ x,
                                                        unsigned short* __restrict__ xb) {
  size_t i = ((size_t)blockIdx.x * blockDim.x + threadIdx.x) * 4;
  f32x4 v = *reinterpret_cast<const f32x4*>(x + i);
  uint32_t lo = (uint32_t)f32_to_bf16_rne(v.x) | ((uint32_t)f32_to_bf16_rne(v.y) << 16);
  uint32_t hi = (uint32_t)f32_to_bf16_rne(v.z) | ((uint32_t)f32_to_bf16_rne(v.w) << 16);
  uint32_t* o = reinterpret_cast<uint32_t*>(xb + i);
  o[0] = lo; o[1] = hi;
}

// ---------------------------------------------------------------------------
// Batched transpose + convert: in [batch][K][N] f32 -> out [batch][N][K] bf16
// grid (N/32, K/32, batch), block (32,8)
// ---------------------------------------------------------------------------
__global__ __launch_bounds__(256) void transpose_bf16_kernel(
    const float* __restrict__ in, unsigned short* __restrict__ out, int K, int N) {
  __shared__ float tile[32][33];
  const float* src = in + (size_t)blockIdx.z * K * N;
  unsigned short* dst = out + (size_t)blockIdx.z * N * K;
  const int n0 = blockIdx.x * 32, k0 = blockIdx.y * 32;
  for (int i = threadIdx.y; i < 32; i += 8)
    tile[i][threadIdx.x] = src[(size_t)(k0 + i) * N + n0 + threadIdx.x];
  __syncthreads();
  for (int i = threadIdx.y; i < 32; i += 8)
    dst[(size_t)(n0 + i) * K + k0 + threadIdx.x] = f32_to_bf16_rne(tile[threadIdx.x][i]);
}

// ---------------------------------------------------------------------------
// GEMM1: for list (e,k): H[slot] = silu(X@Wg) * (X@Wu), bf16 out.
// grid (INTER/64, T/64, NEXP*2), block 128 (4 waves).
// Block tile 64x64, Kstep 32, double-buffered LDS, async g->LDS when available.
// ---------------------------------------------------------------------------
__global__ __launch_bounds__(128) void gemm1_kernel(
    const unsigned short* __restrict__ xb,    // [T][HIDDEN] bf16
    const unsigned short* __restrict__ wgT,   // [E][INTER][HIDDEN] bf16
    const unsigned short* __restrict__ wuT,   // [E][INTER][HIDDEN] bf16
    const int* __restrict__ cnt, const int* __restrict__ toklist,
    unsigned short* __restrict__ hb)          // [2T][INTER] bf16
{
  const int nb = blockIdx.x * 64;
  const int rb = blockIdx.y * 64;
  const int ek = blockIdx.z;
  const int e  = ek >> 1;
  const int kk = ek & 1;
  const int n_tokens = cnt[ek];
  if (rb >= n_tokens) return;
  const int rows = min(64, n_tokens - rb);

  __shared__ int s_tok[64];
  __shared__ alignas(16) unsigned short As[2][64][40];
  __shared__ alignas(16) unsigned short Bg[2][64][40];
  __shared__ alignas(16) unsigned short Bu[2][64][40];

  const int tid = threadIdx.x;
  if (tid < 64) {
    int r = min(rb + tid, n_tokens - 1);
    s_tok[tid] = toklist[(size_t)ek * T_TOKENS + r];
  }
  __syncthreads();

  const int wave = tid >> 5, lane = tid & 31;
  const int wm = (wave >> 1) * 32, wn = (wave & 1) * 32;
  const int lr = tid >> 1, lh = (tid & 1) * 16;

  const unsigned short* gA = xb  + (size_t)s_tok[lr] * HIDDEN + lh;
  const unsigned short* gG = wgT + (size_t)e * INTER * HIDDEN + (size_t)(nb + lr) * HIDDEN + lh;
  const unsigned short* gU = wuT + (size_t)e * INTER * HIDDEN + (size_t)(nb + lr) * HIDDEN + lh;

  auto issue = [&](int k0, int buf) {
    copy16_g2l(gA + k0,     &As[buf][lr][lh]);
    copy16_g2l(gA + k0 + 8, &As[buf][lr][lh + 8]);
    copy16_g2l(gG + k0,     &Bg[buf][lr][lh]);
    copy16_g2l(gG + k0 + 8, &Bg[buf][lr][lh + 8]);
    copy16_g2l(gU + k0,     &Bu[buf][lr][lh]);
    copy16_g2l(gU + k0 + 8, &Bu[buf][lr][lh + 8]);
  };

  v8f accG[2][2], accU[2][2];
#pragma unroll
  for (int i = 0; i < 2; ++i)
#pragma unroll
    for (int j = 0; j < 2; ++j) {
      accG[i][j] = (v8f){0,0,0,0,0,0,0,0};
      accU[i][j] = (v8f){0,0,0,0,0,0,0,0};
    }

  const int krow  = (lane >> 4) * 8;    // A chunks: [krow,krow+8) and [krow+16,krow+24)
  const int kcol  = (lane >> 4) * 16;   // unused; B chunks use krow like A
  const int lanem = lane & 15;

  issue(0, 0);
  wait_async_all();
  __syncthreads();

  const int NT = HIDDEN / 32;
  for (int i = 0; i < NT; ++i) {
    const int cur = i & 1;
    if (i + 1 < NT) issue((i + 1) * 32, 1 - cur);

    v16bf a[2], bg[2], bu[2];
#pragma unroll
    for (int mi = 0; mi < 2; ++mi) {
      int row = wm + mi * 16 + lanem;
      a[mi] = load_frag_lds(&As[cur][row][krow], &As[cur][row][krow + 16]);
    }
#pragma unroll
    for (int ni = 0; ni < 2; ++ni) {
      int n = wn + ni * 16 + lanem;
      bg[ni] = load_frag_lds(&Bg[cur][n][krow], &Bg[cur][n][krow + 16]);
      bu[ni] = load_frag_lds(&Bu[cur][n][krow], &Bu[cur][n][krow + 16]);
    }
#pragma unroll
    for (int mi = 0; mi < 2; ++mi)
#pragma unroll
      for (int ni = 0; ni < 2; ++ni) {
        accG[mi][ni] = WMMA_BF16(a[mi], bg[ni], accG[mi][ni]);
        accU[mi][ni] = WMMA_BF16(a[mi], bu[ni], accU[mi][ni]);
      }

    wait_async_all();
    __syncthreads();
  }

#pragma unroll
  for (int mi = 0; mi < 2; ++mi)
#pragma unroll
    for (int ni = 0; ni < 2; ++ni) {
      int n = nb + wn + ni * 16 + lanem;
#pragma unroll
      for (int r = 0; r < 8; ++r) {
        int row = wm + mi * 16 + ((lane >> 4) * 8) + r;
        if (row < rows) {
          float g = accG[mi][ni][r], u = accU[mi][ni][r];
          float h = g * FAST_RCP(1.0f + __expf(-g)) * u;   // silu(g) * u
          size_t slot = (size_t)s_tok[row] * 2 + kk;
          hb[slot * INTER + n] = f32_to_bf16_rne(h);
        }
      }
    }
}

// ---------------------------------------------------------------------------
// GEMM2: out[tok] (+)= w * (H[slot] @ Wd).  Pass kk=0: store, kk=1: add.
// grid (HIDDEN/128, T/64, NEXP), block 128. Block tile 64x128, wave 32x64.
// ---------------------------------------------------------------------------
__global__ __launch_bounds__(128) void gemm2_kernel(
    const unsigned short* __restrict__ hb,    // [2T][INTER] bf16
    const unsigned short* __restrict__ wdT,   // [E][HIDDEN][INTER] bf16
    const int* __restrict__ cnt, const int* __restrict__ toklist,
    const float* __restrict__ wtlist, float* __restrict__ out,
    int kk, int do_add)
{
  const int nb = blockIdx.x * 128;
  const int rb = blockIdx.y * 64;
  const int e  = blockIdx.z;
  const int ek = e * 2 + kk;
  const int n_tokens = cnt[ek];
  if (rb >= n_tokens) return;
  const int rows = min(64, n_tokens - rb);

  __shared__ int   s_tok[64];
  __shared__ float s_wt[64];
  __shared__ alignas(16) unsigned short As[2][64][40];
  __shared__ alignas(16) unsigned short Bs[2][128][40];

  const int tid = threadIdx.x;
  if (tid < 64) {
    int r = min(rb + tid, n_tokens - 1);
    s_tok[tid] = toklist[(size_t)ek * T_TOKENS + r];
    s_wt[tid]  = wtlist [(size_t)ek * T_TOKENS + r];
  }
  __syncthreads();

  const int wave = tid >> 5, lane = tid & 31;
  const int wm = (wave >> 1) * 32, wn = (wave & 1) * 64;
  const int lrA = tid >> 1, lhA = (tid & 1) * 16;

  const unsigned short* gA = hb + ((size_t)s_tok[lrA] * 2 + kk) * INTER + lhA;
  const unsigned short* gB = wdT + (size_t)e * HIDDEN * INTER + (size_t)(nb + tid) * INTER;

  auto issue = [&](int k0, int buf) {
    copy16_g2l(gA + k0,     &As[buf][lrA][lhA]);
    copy16_g2l(gA + k0 + 8, &As[buf][lrA][lhA + 8]);
#pragma unroll
    for (int c = 0; c < 32; c += 8)
      copy16_g2l(gB + k0 + c, &Bs[buf][tid][c]);
  };

  v8f acc[2][4];
#pragma unroll
  for (int i = 0; i < 2; ++i)
#pragma unroll
    for (int j = 0; j < 4; ++j) acc[i][j] = (v8f){0,0,0,0,0,0,0,0};

  const int krow  = (lane >> 4) * 8;
  const int kcol  = (lane >> 4) * 16;
  const int lanem = lane & 15;

  issue(0, 0);
  wait_async_all();
  __syncthreads();

  const int NT = INTER / 32;
  for (int i = 0; i < NT; ++i) {
    const int cur = i & 1;
    if (i + 1 < NT) issue((i + 1) * 32, 1 - cur);

    v16bf a[2], b[4];
#pragma unroll
    for (int mi = 0; mi < 2; ++mi) {
      int row = wm + mi * 16 + lanem;
      a[mi] = load_frag_lds(&As[cur][row][krow], &As[cur][row][krow + 16]);
    }
#pragma unroll
    for (int ni = 0; ni < 4; ++ni) {
      int n = wn + ni * 16 + lanem;
      b[ni] = load_frag_lds(&Bs[cur][n][krow], &Bs[cur][n][krow + 16]);
    }
#pragma unroll
    for (int mi = 0; mi < 2; ++mi)
#pragma unroll
      for (int ni = 0; ni < 4; ++ni)
        acc[mi][ni] = WMMA_BF16(a[mi], b[ni], acc[mi][ni]);

    wait_async_all();
    __syncthreads();
  }

#pragma unroll
  for (int mi = 0; mi < 2; ++mi)
#pragma unroll
    for (int ni = 0; ni < 4; ++ni) {
      int n = nb + wn + ni * 16 + lanem;
#pragma unroll
      for (int r = 0; r < 8; ++r) {
        int row = wm + mi * 16 + ((lane >> 4) * 8) + r;
        if (row < rows) {
          float val = acc[mi][ni][r] * s_wt[row];
          size_t o = (size_t)s_tok[row] * HIDDEN + n;
          if (do_add) out[o] += val;
          else        out[o]  = val;
        }
      }
    }
}

// ---------------------------------------------------------------------------
extern "C" void kernel_launch(void* const* d_in, const int* in_sizes, int n_in,
                              void* d_out, int out_size, void* d_ws, size_t ws_size,
                              hipStream_t stream) {
  (void)in_sizes; (void)n_in; (void)out_size; (void)ws_size;
  const float* x     = (const float*)d_in[0];   // [B,S,HIDDEN]
  const float* Wgate = (const float*)d_in[1];   // [HIDDEN, NEXP]
  const float* Wg    = (const float*)d_in[2];   // [NEXP, HIDDEN, INTER]
  const float* Wu    = (const float*)d_in[3];   // [NEXP, HIDDEN, INTER]
  const float* Wd    = (const float*)d_in[4];   // [NEXP, INTER, HIDDEN]

  float* out_final  = (float*)d_out;                                  // [T, HIDDEN]
  float* logits_out = (float*)d_out + (size_t)T_TOKENS * HIDDEN;      // [T, NEXP]

  char* p = (char*)d_ws;
  auto take = [&](size_t bytes) {
    char* r = p;
    p += (bytes + 255) & ~(size_t)255;
    return r;
  };
  unsigned short* xb  = (unsigned short*)take((size_t)T_TOKENS * HIDDEN * 2);
  unsigned short* wgT = (unsigned short*)take((size_t)NEXP * INTER * HIDDEN * 2);
  unsigned short* wuT = (unsigned short*)take((size_t)NEXP * INTER * HIDDEN * 2);
  unsigned short* wdT = (unsigned short*)take((size_t)NEXP * HIDDEN * INTER * 2);
  unsigned short* hb  = (unsigned short*)take((size_t)T_TOKENS * 2 * INTER * 2);
  int*   cnt     = (int*)  take(NEXP * 2 * sizeof(int));
  int*   toklist = (int*)  take((size_t)NEXP * 2 * T_TOKENS * sizeof(int));
  float* wtlist  = (float*)take((size_t)NEXP * 2 * T_TOKENS * sizeof(float));

  zero_cnt_kernel<<<1, 64, 0, stream>>>(cnt);
  router_kernel<<<T_TOKENS / 8, 256, 0, stream>>>(x, Wgate, logits_out, cnt, toklist, wtlist);
  convert_x_kernel<<<(T_TOKENS * HIDDEN) / (256 * 4), 256, 0, stream>>>(x, xb);
  transpose_bf16_kernel<<<dim3(INTER / 32, HIDDEN / 32, NEXP), dim3(32, 8), 0, stream>>>(Wg, wgT, HIDDEN, INTER);
  transpose_bf16_kernel<<<dim3(INTER / 32, HIDDEN / 32, NEXP), dim3(32, 8), 0, stream>>>(Wu, wuT, HIDDEN, INTER);
  transpose_bf16_kernel<<<dim3(HIDDEN / 32, INTER / 32, NEXP), dim3(32, 8), 0, stream>>>(Wd, wdT, INTER, HIDDEN);
  gemm1_kernel<<<dim3(INTER / 64, T_TOKENS / 64, NEXP * 2), 128, 0, stream>>>(xb, wgT, wuT, cnt, toklist, hb);
  gemm2_kernel<<<dim3(HIDDEN / 128, T_TOKENS / 64, NEXP), 128, 0, stream>>>(hb, wdT, cnt, toklist, wtlist, out_final, 0, 0);
  gemm2_kernel<<<dim3(HIDDEN / 128, T_TOKENS / 64, NEXP), 128, 0, stream>>>(hb, wdT, cnt, toklist, wtlist, out_final, 1, 1);
}
